// CausalSelfAttention_46720654246234
// MI455X (gfx1250) — matrix operand order repaired and checked by reading, not compiled, not run on hardware
//
#include <hip/hip_runtime.h>

#define B_  4
#define L_  2048
#define D_  1024
#define H_  16
#define DH_ 64

#define SROW  2052  // padded LDS row stride (floats) for attention strip
#define BPITCH 56   // padded LDS row pitch (u16) for staged B tiles: 112B, 16B-aligned, conflict-free

typedef __attribute__((ext_vector_type(16))) __bf16 v16bf;
typedef __attribute__((ext_vector_type(8)))  float  v8f;
typedef unsigned int  u32;
typedef unsigned short u16;
typedef __attribute__((ext_vector_type(4))) u32   u32x4;
typedef __attribute__((ext_vector_type(2))) float f32x2;
typedef __attribute__((ext_vector_type(4))) float f32x4;

#define NEG_INF (-__builtin_huge_valf())

union FragU {
  v16bf v;
  u16   h[16];
  u32   w[8];
  u32x4 q[2];
};

__device__ __forceinline__ u16 f2bf(float f) {
  u32 u = __float_as_uint(f);
  u32 r = u + 0x7FFFu + ((u >> 16) & 1u);   // round-to-nearest-even
  return (u16)(r >> 16);
}

// A fragment: 16x32 bf16 row-major = two contiguous 16B chunks per lane.
// lane l: m=l%16, half=l/16; elements e<8: k=half*8+e, e>=8: k=16+half*8+(e-8).
__device__ __forceinline__ v16bf load_a(const u16* A, size_t row, int ld, int kbase, int lane) {
  int m = lane & 15, half = lane >> 4;
  const u16* base = A + (row + (size_t)m) * (size_t)ld + kbase + half * 8;
  FragU f;
  f.q[0] = *(const u32x4*)(base);
  f.q[1] = *(const u32x4*)(base + 16);
  return f.v;
}

// A fragment from fp32 LDS rows (attention probabilities), converted to bf16.
__device__ __forceinline__ v16bf load_a_lds(const float* S, int kbase, int lane) {
  int m = lane & 15, half = lane >> 4;
  const float* base = S + m * SROW + kbase;
  FragU f;
#pragma unroll
  for (int i = 0; i < 8; ++i) {
    int j = i & 3, g = i >> 2;
    int k = g * 16 + half * 8 + 2 * j;
    f32x2 w = *(const f32x2*)(base + k);
    f.w[i] = (u32)f2bf(w.x) | ((u32)f2bf(w.y) << 16);
  }
  return f.v;
}

// B fragment from global, B stored N-major (Bt[n][k]); element e<8 <-> K = khalf*8 + e, e>=8 <-> K = 16 + khalf*8 + (e-8).
__device__ __forceinline__ v16bf load_bT(const u16* Bt, size_t col, int ld, int kbase, int lane) {
  int n = lane & 15, kh = lane >> 4;
  const u16* p = Bt + (col + (size_t)n) * (size_t)ld + kbase + kh * 8;
  FragU f;
  f.q[0] = *(const u32x4*)(p);
  f.q[1] = *(const u32x4*)(p + 16);
  return f.v;
}

// B fragment from an LDS-staged 64x32 tile with BPITCH row pitch.
__device__ __forceinline__ v16bf load_b_lds(const u16* Bl, int col0, int lane) {
  int n = lane & 15, kh = lane >> 4;
  const u16* p = Bl + (col0 + n) * BPITCH + kh * 8;
  FragU f;
  f.q[0] = *(const u32x4*)(p);
  f.q[1] = *(const u32x4*)(p + 16);
  return f.v;
}

__device__ __forceinline__ void zero_acc(v8f& a) {
#pragma unroll
  for (int r = 0; r < 8; ++r) a[r] = 0.0f;
}

// Asynchronously copy a 64-col x 32-k bf16 B tile (4KB) into LDS.
// 128 threads x 2 chunks x 16B. Tracked by ASYNCcnt (2 per thread per stage).
__device__ __forceinline__ void copy_b_async(u16* dst, const u16* Bt, int col0, int kb,
                                             int ldb, int tid) {
#pragma unroll
  for (int j = 0; j < 2; ++j) {
    int chunk = tid * 2 + j;                 // 0..255
    int c = chunk >> 2, part = chunk & 3;    // row in tile, 16B part
    const u16* g = Bt + (size_t)(col0 + c) * (size_t)ldb + kb + part * 8;
    unsigned lds = (unsigned)(unsigned long long)(const void*)(dst + c * BPITCH + part * 8);
    asm volatile("global_load_async_to_lds_b128 %0, %1, off"
                 :: "v"(lds), "v"(g) : "memory");
  }
}

// Block-cooperative GEMM: 4 waves x (16 rows x 64 cols), B tile double-buffered in LDS
// via async-to-LDS; A fragments double-buffered in registers.
__device__ __forceinline__ void gemm_block_16x64(const u16* A, size_t arow, int lda,
                                                 const u16* Bt, int col0, int ldb,
                                                 int K, u16* Bsh, int tid, v8f acc[4]) {
  int lane = tid & 31;
  int nstage = K / 32;
  copy_b_async(Bsh, Bt, col0, 0, ldb, tid);
  v16bf a_cur = load_a(A, arow, lda, 0, lane);
#pragma unroll 2
  for (int s = 0; s < nstage; ++s) {
    const u16* Bl = Bsh + (s & 1) * (64 * BPITCH);
    v16bf a_nxt = a_cur;
    if (s + 1 < nstage) {
      copy_b_async(Bsh + ((s + 1) & 1) * (64 * BPITCH), Bt, col0, (s + 1) * 32, ldb, tid);
      a_nxt = load_a(A, arow, lda, (s + 1) * 32, lane);
      asm volatile("s_wait_asynccnt 0x2" ::: "memory");   // previous stage's copies done
    } else {
      asm volatile("s_wait_asynccnt 0x0" ::: "memory");
    }
    __syncthreads();                                       // tile visible to all waves
#pragma unroll
    for (int t = 0; t < 4; ++t) {
      v16bf b = load_b_lds(Bl, 16 * t, lane);
      acc[t] = __builtin_amdgcn_wmma_f32_16x16x32_bf16(false, a_cur, false, b,
                                                       (short)0, acc[t], false, false);
    }
    __syncthreads();                                       // all reads done before overwrite
    a_cur = a_nxt;
  }
}

// ---------------- conversion kernels ----------------

__global__ void cvt_bf16(const float* __restrict__ x, u16* __restrict__ y, size_t n) {
  size_t i = (size_t)blockIdx.x * 256 + threadIdx.x;
  if (i < n) y[i] = f2bf(x[i]);
}

__global__ void cvt_transpose_bf16(const float* __restrict__ w, u16* __restrict__ wT,
                                   int K, int N) {
  size_t i = (size_t)blockIdx.x * 256 + threadIdx.x;
  if (i >= (size_t)K * N) return;
  int k = (int)(i / N), n = (int)(i % N);
  wT[(size_t)n * K + k] = f2bf(w[i]);
}

// ---------------- QKV projection ----------------
__global__ __launch_bounds__(128) void qkv_gemm(const u16* __restrict__ xb,
                                                const u16* __restrict__ wT,
                                                u16* __restrict__ qh,
                                                u16* __restrict__ kh,
                                                u16* __restrict__ vT) {
  __shared__ u16 Bsh[2 * 64 * BPITCH];
  int tid = threadIdx.x;
  int lane = tid & 31, wave = tid >> 5;
  size_t row = (size_t)blockIdx.x * 64 + (size_t)wave * 16;
  int col = blockIdx.y * 64;
  v8f acc[4];
#pragma unroll
  for (int t = 0; t < 4; ++t) zero_acc(acc[t]);

  gemm_block_16x64(xb, row, D_, wT, col, D_, D_, Bsh, tid, acc);

  int n = lane & 15, mh = lane >> 4;
  int sec = col >> 10;
  int h = (col >> 6) & 15;
  int b = (int)(row >> 11);
  int lbase = (int)(row & 2047);
  size_t bh = (size_t)b * H_ + h;
#pragma unroll
  for (int t = 0; t < 4; ++t) {
#pragma unroll
    for (int r = 0; r < 8; ++r) {
      int l = lbase + r + 8 * mh;
      int dh = 16 * t + n;
      u16 val = f2bf(acc[t][r]);
      if (sec == 0)      qh[(bh * L_ + l) * DH_ + dh] = val;
      else if (sec == 1) kh[(bh * L_ + l) * DH_ + dh] = val;
      else               vT[(bh * DH_ + dh) * L_ + l] = val;
    }
  }
}

// ---------------- fused attention: scores -> softmax -> PV + attn write-out ----------------
__global__ __launch_bounds__(128) void attn_fused(const u16* __restrict__ qh,
                                                  const u16* __restrict__ kh,
                                                  const unsigned char* __restrict__ pad,
                                                  const u16* __restrict__ vT,
                                                  float* __restrict__ attn,
                                                  u16* __restrict__ obf) {
  extern __shared__ float smem[];           // S[16][SROW] + red[16][8]
  float* S   = smem;
  float* red = smem + 16 * SROW;

  int tid = threadIdx.x;
  int lane = tid & 31, wave = tid >> 5;
  int bhid = blockIdx.x;
  int qt   = blockIdx.y;
  int b = bhid >> 4, h = bhid & 15;
  int n = lane & 15, mh = lane >> 4;

  const u16* qbh = qh + (size_t)bhid * L_ * DH_;
  const u16* kbh = kh + (size_t)bhid * L_ * DH_;
  const u16* vbh = vT + (size_t)bhid * DH_ * L_;

  int kend = (qt + 1) * 16;
  int kmax = (kend + 31) & ~31;

  if (kmax > kend) {                        // zero the PV pad tile [kend, kmax)
    int r = tid >> 3;
    int c = kend + (tid & 7) * 2;
    S[r * SROW + c] = 0.f;
    S[r * SROW + c + 1] = 0.f;
  }

  // ---- scores: wave w handles k-tiles w, w+4, ... <= qt ----
  v16bf aq0 = load_a(qbh, (size_t)qt * 16, DH_, 0,  lane);
  v16bf aq1 = load_a(qbh, (size_t)qt * 16, DH_, 32, lane);
  for (int kt = wave; kt <= qt; kt += 4) {
    v16bf b0 = load_bT(kbh, (size_t)kt * 16, DH_, 0,  lane);
    v16bf b1 = load_bT(kbh, (size_t)kt * 16, DH_, 32, lane);
    v8f acc; zero_acc(acc);
    acc = __builtin_amdgcn_wmma_f32_16x16x32_bf16(false, aq0, false, b0, (short)0, acc, false, false);
    acc = __builtin_amdgcn_wmma_f32_16x16x32_bf16(false, aq1, false, b1, (short)0, acc, false, false);
#pragma unroll
    for (int r = 0; r < 8; ++r) {
      int qi = qt * 16 + r + 8 * mh;
      int ki = kt * 16 + n;
      float v = acc[r] * 0.125f;
      if (ki > qi || pad[b * L_ + ki]) v = NEG_INF;
      S[(r + 8 * mh) * SROW + kt * 16 + n] = v;
    }
  }
  __syncthreads();

  // ---- softmax over LDS rows: 8 threads per row ----
  {
    int r  = tid >> 3;
    int t8 = tid & 7;
    float* row = S + r * SROW;

    float mx = NEG_INF;
    for (int c = t8 * 4; c < kend; c += 32) {
      f32x4 v = *(const f32x4*)(row + c);
      mx = fmaxf(mx, fmaxf(fmaxf(v.x, v.y), fmaxf(v.z, v.w)));
    }
    red[r * 8 + t8] = mx;
    __syncthreads();
    float m = red[r * 8 + 0];
#pragma unroll
    for (int i = 1; i < 8; ++i) m = fmaxf(m, red[r * 8 + i]);
    if (m < -3.0e38f) m = 0.f;            // fully-masked row
    __syncthreads();

    float sum = 0.f;
    for (int c = t8 * 4; c < kend; c += 32) {
      f32x4 v = *(const f32x4*)(row + c);
      f32x4 e = {__expf(v.x - m), __expf(v.y - m), __expf(v.z - m), __expf(v.w - m)};
      *(f32x4*)(row + c) = e;
      sum += e.x + e.y + e.z + e.w;
    }
    red[r * 8 + t8] = sum;
    __syncthreads();
    float tot = 0.f;
#pragma unroll
    for (int i = 0; i < 8; ++i) tot += red[r * 8 + i];
    float inv = (tot > 0.f) ? 1.0f / tot : 0.f;   // nan_to_num
    for (int c = t8 * 4; c < kend; c += 32) {
      f32x4 v = *(const f32x4*)(row + c);
      f32x4 o = {v.x * inv, v.y * inv, v.z * inv, v.w * inv};
      *(f32x4*)(row + c) = o;
    }
  }
  __syncthreads();

  // ---- stream attention weights to HBM (written exactly once) ----
  {
    float* out = attn + ((size_t)bhid * L_ + (size_t)qt * 16) * L_;
    f32x4 z = {0.f, 0.f, 0.f, 0.f};
#pragma unroll 1
    for (int r = 0; r < 16; ++r) {
      for (int c = tid * 4; c < L_; c += 512) {
        f32x4 v = (c < kmax) ? *(const f32x4*)(S + r * SROW + c) : z;
        *(f32x4*)(out + (size_t)r * L_ + c) = v;
      }
    }
  }

  // ---- PV: wave w computes d-tile t=w, P read straight from LDS ----
  {
    v8f acc; zero_acc(acc);
    for (int k = 0; k < kmax; k += 32) {
      v16bf a  = load_a_lds(S, k, lane);
      v16bf bb = load_bT(vbh, (size_t)(16 * wave), L_, k, lane);
      acc = __builtin_amdgcn_wmma_f32_16x16x32_bf16(false, a, false, bb, (short)0, acc, false, false);
    }
#pragma unroll
    for (int r = 0; r < 8; ++r) {
      int q = qt * 16 + r + 8 * mh;
      obf[((size_t)(b * L_ + q)) * D_ + h * DH_ + wave * 16 + n] = f2bf(acc[r]);
    }
  }
}

// ---------------- output projection ----------------
__global__ __launch_bounds__(128) void out_gemm(const u16* __restrict__ obf,
                                                const u16* __restrict__ wT,
                                                float* __restrict__ out) {
  __shared__ u16 Bsh[2 * 64 * BPITCH];
  int tid = threadIdx.x;
  int lane = tid & 31, wave = tid >> 5;
  size_t row = (size_t)blockIdx.x * 64 + (size_t)wave * 16;
  int col = blockIdx.y * 64;
  v8f acc[4];
#pragma unroll
  for (int t = 0; t < 4; ++t) zero_acc(acc[t]);

  gemm_block_16x64(obf, row, D_, wT, col, D_, D_, Bsh, tid, acc);

  int n = lane & 15, mh = lane >> 4;
#pragma unroll
  for (int t = 0; t < 4; ++t) {
#pragma unroll
    for (int r = 0; r < 8; ++r)
      out[(row + r + 8 * mh) * (size_t)D_ + col + 16 * t + n] = acc[t][r];
  }
}

extern "C" void kernel_launch(void* const* d_in, const int* in_sizes, int n_in,
                              void* d_out, int out_size, void* d_ws, size_t ws_size,
                              hipStream_t stream) {
  const float* x           = (const float*)d_in[0];
  const unsigned char* pad = (const unsigned char*)d_in[1];
  const float* w_qkv       = (const float*)d_in[2];
  const float* w_out       = (const float*)d_in[3];

  float* out  = (float*)d_out;
  float* attn = out + (size_t)B_ * L_ * D_;

  u16* xb    = (u16*)d_ws;
  u16* wqkvT = xb    + (size_t)B_ * L_ * D_;
  u16* woutT = wqkvT + (size_t)3 * D_ * D_;
  u16* qh    = woutT + (size_t)D_ * D_;
  u16* kh    = qh    + (size_t)B_ * H_ * L_ * DH_;
  u16* vT    = kh    + (size_t)B_ * H_ * L_ * DH_;
  u16* obf   = vT    + (size_t)B_ * H_ * L_ * DH_;

  size_t nx = (size_t)B_ * L_ * D_;
  cvt_bf16<<<dim3((unsigned)((nx + 255) / 256)), 256, 0, stream>>>(x, xb, nx);
  cvt_transpose_bf16<<<dim3((3 * D_ * D_ + 255) / 256), 256, 0, stream>>>(w_qkv, wqkvT, D_, 3 * D_);
  cvt_transpose_bf16<<<dim3((D_ * D_ + 255) / 256), 256, 0, stream>>>(w_out, woutT, D_, D_);

  qkv_gemm<<<dim3(128, 48), 128, 0, stream>>>(xb, wqkvT, qh, kh, vT);

  size_t lds_bytes = (size_t)(16 * SROW + 16 * 8) * sizeof(float);  // 131,840 B
  attn_fused<<<dim3(B_ * H_, L_ / 16), 128, lds_bytes, stream>>>(qh, kh, pad, vT, attn, obf);

  out_gemm<<<dim3(128, 16), 128, 0, stream>>>(obf, woutT, out);
}
